// LSTMAutoEncoder_19121194401961
// MI455X (gfx1250) — matrix operand order repaired and checked by reading, not compiled, not run on hardware
//
#include <hip/hip_runtime.h>

// ---------------------------------------------------------------------------
// LSTM seq2seq autoencoder for MI455X (gfx1250, wave32).
// B=512, T=512, D=128, L=256.  Persistent-kernel recurrence with
// v_wmma_f32_16x16x32_bf16, LDS-staged weights, in-register gate fusion.
// ---------------------------------------------------------------------------

typedef __attribute__((ext_vector_type(16))) __bf16 v16bf;
typedef __attribute__((ext_vector_type(8)))  float  v8f;

#define kB 512
#define kT 512
#define kD 128
#define kL 256
#define NWG 64          // persistent workgroups (512 waves total)

__device__ __forceinline__ unsigned short f2bf(float f) {
  union { __bf16 h; unsigned short u; } x;
  x.h = (__bf16)f;                       // native v_cvt (RNE) on gfx1250
  return x.u;
}

union Frag {
  v16bf v;
  unsigned short us[16];
  uint4 q[2];
};

__device__ __forceinline__ v8f wmma_bf16(const Frag& a, const Frag& b, v8f c) {
  return __builtin_amdgcn_wmma_f32_16x16x32_bf16(
      /*neg_a=*/false, a.v, /*neg_b=*/false, b.v,
      /*c_mod=*/(short)0, c, /*reuse_a=*/false, /*reuse_b=*/false);
}

// A fragment (16x32 bf16): lane<16 -> row, K groups {0..7,16..23};
// lane>=16 -> K groups {8..15,24..31}. Row-major bf16 source (global).
__device__ __forceinline__ void loadA_bf(Frag& a, const unsigned short* row,
                                         int kb, bool hi) {
  const unsigned short* p = row + kb + (hi ? 8 : 0);
  a.q[0] = *(const uint4*)p;
  a.q[1] = *(const uint4*)(p + 16);
}

// Same A fragment from fp32 source, converted in-register to bf16.
__device__ __forceinline__ void loadA_f32(Frag& a, const float* row,
                                          int kb, bool hi) {
  const float* p = row + kb + (hi ? 8 : 0);
  float4 f0 = *(const float4*)p;
  float4 f1 = *(const float4*)(p + 4);
  float4 f2 = *(const float4*)(p + 16);
  float4 f3 = *(const float4*)(p + 20);
  a.us[0]=f2bf(f0.x);  a.us[1]=f2bf(f0.y);  a.us[2]=f2bf(f0.z);  a.us[3]=f2bf(f0.w);
  a.us[4]=f2bf(f1.x);  a.us[5]=f2bf(f1.y);  a.us[6]=f2bf(f1.z);  a.us[7]=f2bf(f1.w);
  a.us[8]=f2bf(f2.x);  a.us[9]=f2bf(f2.y);  a.us[10]=f2bf(f2.z); a.us[11]=f2bf(f2.w);
  a.us[12]=f2bf(f3.x); a.us[13]=f2bf(f3.y); a.us[14]=f2bf(f3.z); a.us[15]=f2bf(f3.w);
}

// B fragment (32x16 bf16) from LDS-staged transposed weights: local column
// q*16+j, K contiguous.  lane<16 -> K {0..7,16..23}, lane>=16 -> K {8..15,24..31}.
__device__ __forceinline__ void loadB_s(Frag& b, const unsigned short* sw,
                                        int K, int q, int j, int kb, bool hi) {
  const unsigned short* p = sw + (q * 16 + j) * K + kb + (hi ? 8 : 0);
  b.q[0] = *(const uint4*)p;
  b.q[1] = *(const uint4*)(p + 16);
}

// acc[q] accumulates gate q's 16x16 tile.
__device__ __forceinline__ void gemm4_bf(v8f acc[4], const unsigned short* A,
                                         int arow, int myrow,
                                         const unsigned short* sW, int K,
                                         int j, bool hi) {
  const unsigned short* row = A + myrow * arow;
  for (int kb = 0; kb < K; kb += 32) {
    Frag a; loadA_bf(a, row, kb, hi);
#pragma unroll
    for (int q = 0; q < 4; ++q) {
      Frag b; loadB_s(b, sW, K, q, j, kb, hi);
      acc[q] = wmma_bf16(a, b, acc[q]);
    }
  }
}

__device__ __forceinline__ void gemm4_f32(v8f acc[4], const float* xrow,
                                          const unsigned short* sW, int K,
                                          int j, bool hi) {
  for (int kb = 0; kb < K; kb += 32) {
    Frag a; loadA_f32(a, xrow, kb, hi);
#pragma unroll
    for (int q = 0; q < 4; ++q) {
      Frag b; loadB_s(b, sW, K, q, j, kb, hi);
      acc[q] = wmma_bf16(a, b, acc[q]);
    }
  }
}

__device__ __forceinline__ float sigmf(float z) {
  return 1.0f / (1.0f + __expf(-z));
}

// In-register LSTM gate fusion: acc[0..3] = zi,zf,zg,zo for one 16x16 patch.
__device__ __forceinline__ void lstm_update(v8f acc[4], const float* bias,
                                            float* c, unsigned short* hbf,
                                            int row0, int colL, bool hi) {
  float bi = bias[colL];
  float bf_ = bias[kL + colL];
  float bg = bias[2 * kL + colL];
  float bo = bias[3 * kL + colL];
  int rbase = row0 + (hi ? 8 : 0);
#pragma unroll
  for (int e = 0; e < 8; ++e) {
    int addr = (rbase + e) * kL + colL;
    float i_ = sigmf(acc[0][e] + bi);
    float f_ = sigmf(acc[1][e] + bf_);
    float g_ = fmaxf(acc[2][e] + bg, 0.0f);
    float o_ = sigmf(acc[3][e] + bo);
    float cn = f_ * c[addr] + i_ * g_;
    c[addr] = cn;
    hbf[addr] = f2bf(o_ * fmaxf(cn, 0.0f));
  }
}

// Grid-wide split barrier via device-scope atomics; counters reset by prep.
__device__ __forceinline__ void gbar(unsigned* cnt, unsigned* gen) {
  __syncthreads();
  if (threadIdx.x == 0) {
    unsigned g = __hip_atomic_load(gen, __ATOMIC_RELAXED, __HIP_MEMORY_SCOPE_AGENT);
    unsigned prev = __hip_atomic_fetch_add(cnt, 1u, __ATOMIC_ACQ_REL,
                                           __HIP_MEMORY_SCOPE_AGENT);
    if (prev == (unsigned)NWG - 1u) {
      __hip_atomic_store(cnt, 0u, __ATOMIC_RELAXED, __HIP_MEMORY_SCOPE_AGENT);
      __hip_atomic_fetch_add(gen, 1u, __ATOMIC_ACQ_REL, __HIP_MEMORY_SCOPE_AGENT);
    } else {
      while (__hip_atomic_load(gen, __ATOMIC_ACQUIRE,
                               __HIP_MEMORY_SCOPE_AGENT) == g) {
        __builtin_amdgcn_s_sleep(2);
      }
    }
  }
  __syncthreads();
}

// Stage this block's 64 weight columns (4 gates x 16 cols for group g) from
// the transposed global bf16 weights into LDS.  uint4 granularity:
// global_load_b128 -> ds_store_b128.
__device__ __forceinline__ void stage64(unsigned short* dst,
                                        const unsigned short* src, int K,
                                        int g) {
  int vpc = K / 8;                       // uint4 per column
  for (int i = threadIdx.x; i < 64 * vpc; i += 256) {
    int col = i / vpc, v = i % vpc;
    int gcol = (col >> 4) * kL + g * 16 + (col & 15);
    *(uint4*)(dst + col * K + v * 8) = *(const uint4*)(src + gcol * K + v * 8);
  }
}

__device__ __forceinline__ void stage16(unsigned short* dst,
                                        const unsigned short* src, int K,
                                        int g) {                 // out_W, g<8
  int vpc = K / 8;
  for (int i = threadIdx.x; i < 16 * vpc; i += 256) {
    int col = i / vpc, v = i % vpc;
    int gcol = g * 16 + col;
    *(uint4*)(dst + col * K + v * 8) = *(const uint4*)(src + gcol * K + v * 8);
  }
}

// ---------------------------------------------------------------------------
// Prep: bf16+transpose weights, zero states, seed decoder input, reset barrier
// ---------------------------------------------------------------------------
__global__ void lstm_prep(const float* W0, const float* U0, const float* W1,
                          const float* U1, const float* dW, const float* dU,
                          const float* oW, const float* dec_in,
                          unsigned short* w0t, unsigned short* u0t,
                          unsigned short* w1t, unsigned short* u1t,
                          unsigned short* dwt, unsigned short* dut,
                          unsigned short* owt, float* c0, float* c1, float* cd,
                          unsigned short* h0, unsigned short* h1,
                          unsigned short* hd, unsigned short* dinp,
                          unsigned* bar) {
  const long N128 = 1024L * 128, N256 = 1024L * 256;
  const long TOTAL = 2 * N128 + 4 * N256 + 32768 + 3 * 131072 + 3 * 131072 + 65536 + 2;
  long stride = (long)gridDim.x * blockDim.x;
  for (long idx = (long)blockIdx.x * blockDim.x + threadIdx.x; idx < TOTAL;
       idx += stride) {
    long r = idx;
    if (r < N128) { long n = r / 128, k = r % 128; w0t[r] = f2bf(W0[k * 1024 + n]); continue; }
    r -= N128;
    if (r < N256) { long n = r / 256, k = r % 256; u0t[r] = f2bf(U0[k * 1024 + n]); continue; }
    r -= N256;
    if (r < N256) { long n = r / 256, k = r % 256; w1t[r] = f2bf(W1[k * 1024 + n]); continue; }
    r -= N256;
    if (r < N256) { long n = r / 256, k = r % 256; u1t[r] = f2bf(U1[k * 1024 + n]); continue; }
    r -= N256;
    if (r < N128) { long n = r / 128, k = r % 128; dwt[r] = f2bf(dW[k * 1024 + n]); continue; }
    r -= N128;
    if (r < N256) { long n = r / 256, k = r % 256; dut[r] = f2bf(dU[k * 1024 + n]); continue; }
    r -= N256;
    if (r < 32768) { long n = r / 256, k = r % 256; owt[r] = f2bf(oW[k * 128 + n]); continue; }
    r -= 32768;
    if (r < 3 * 131072) {
      float* cs = (r < 131072) ? c0 : (r < 262144) ? c1 : cd;
      cs[r % 131072] = 0.0f; continue;
    }
    r -= 3 * 131072;
    if (r < 3 * 131072) {
      unsigned short* hs = (r < 131072) ? h0 : (r < 262144) ? h1 : hd;
      hs[r % 131072] = 0; continue;
    }
    r -= 3 * 131072;
    if (r < 65536) {  // dinp = bf16(decoder_inputs[:,0,:])
      long b = r / 128, d = r % 128;
      dinp[r] = f2bf(dec_in[b * (long)(kT * kD) + d]); continue;
    }
    r -= 65536;
    bar[r] = 0;
  }
}

// ---------------------------------------------------------------------------
// Persistent recurrence kernel: 64 WGs x 256 thr = 512 waves = 32 row-tiles
// x 16 gate-column groups.  Each wave owns one 16x16 (row,l) patch of all 4
// gates.  B-operand weights live in LDS (112 KB encoder / 56 KB decoder,
// overlaid and refilled between phases).
// ---------------------------------------------------------------------------
__global__ void __launch_bounds__(256, 1)
lstm_persist(const float* enc_in, const float* b0, const float* b1,
             const float* db, const float* ob, const unsigned short* w0t,
             const unsigned short* u0t, const unsigned short* w1t,
             const unsigned short* u1t, const unsigned short* dwt,
             const unsigned short* dut, const unsigned short* owt, float* c0,
             float* c1, float* cd, unsigned short* h0, unsigned short* h1,
             unsigned short* hd, unsigned short* dinp, unsigned* bar,
             float* out) {
  __shared__ __align__(16) unsigned short smem[57344];   // 112 KB

  const int lane = threadIdx.x & 31;
  const bool hi = lane >= 16;
  const int j = lane & 15;
  const int wave = (blockIdx.x << 3) | (threadIdx.x >> 5);  // 0..511
  const int rtile = wave & 31;                              // 32 row tiles
  const int g = wave >> 5;                                  // 16 col groups
  const int row0 = rtile << 4;
  const int myrow = row0 + j;
  const int colL = (g << 4) + j;                            // 0..255 in-gate col
  unsigned* cnt = bar;
  unsigned* gen = bar + 1;

  // ---- stage encoder weights into LDS ----
  unsigned short* sW0 = smem;            // 64 x 128
  unsigned short* sU0 = smem + 8192;     // 64 x 256
  unsigned short* sW1 = smem + 24576;    // 64 x 256
  unsigned short* sU1 = smem + 40960;    // 64 x 256
  stage64(sW0, w0t, kD, g);
  stage64(sU0, u0t, kL, g);
  stage64(sW1, w1t, kL, g);
  stage64(sU1, u1t, kL, g);
  __syncthreads();

  // ---------------- encoder ----------------
  for (int t = 0; t < kT; ++t) {
    const float* xrow = enc_in + (myrow * kT + t) * kD;
    if (t + 1 < kT)                           // hide HBM latency for x(t+1)
      __builtin_prefetch(xrow + kD, 0, 3);

    // layer 0: z = x_t @ W0 + h0 @ U0 + b0
    v8f acc[4] = {};
    gemm4_f32(acc, xrow, sW0, kD, j, hi);
    gemm4_bf(acc, h0, kL, myrow, sU0, kL, j, hi);
    lstm_update(acc, b0, c0, h0, row0, colL, hi);
    gbar(cnt, gen);

    // layer 1: z = h0 @ W1 + h1 @ U1 + b1
    v8f acc2[4] = {};
    gemm4_bf(acc2, h0, kL, myrow, sW1, kL, j, hi);
    gemm4_bf(acc2, h1, kL, myrow, sU1, kL, j, hi);
    lstm_update(acc2, b1, c1, h1, row0, colL, hi);
    gbar(cnt, gen);
  }

  // ---- re-stage decoder weights into the same LDS ----
  unsigned short* sDW = smem;            // 64 x 128
  unsigned short* sDU = smem + 8192;     // 64 x 256
  unsigned short* sOW = smem + 24576;    // 16 x 256
  __syncthreads();                       // everyone done reading encoder LDS
  stage64(sDW, dwt, kD, g);
  stage64(sDU, dut, kL, g);
  if (g < 8) stage16(sOW, owt, kL, g);   // block-uniform predicate
  __syncthreads();

  // ---------------- decoder ----------------
  for (int t = 0; t < kT; ++t) {
    v8f acc[4] = {};
    gemm4_bf(acc, dinp, kD, myrow, sDW, kD, j, hi);
    gemm4_bf(acc, hd, kL, myrow, sDU, kL, j, hi);
    lstm_update(acc, db, cd, hd, row0, colL, hi);
    gbar(cnt, gen);

    // out = hd @ out_W + out_b   (512x128, waves with g<8 cover it)
    if (g < 8) {
      v8f o = {};
      const unsigned short* row = hd + myrow * kL;
      for (int kb = 0; kb < kL; kb += 32) {
        Frag a; loadA_bf(a, row, kb, hi);
        Frag b;
        const unsigned short* p = sOW + j * kL + kb + (hi ? 8 : 0);
        b.q[0] = *(const uint4*)p;
        b.q[1] = *(const uint4*)(p + 16);
        o = wmma_bf16(a, b, o);
      }
      float bias = ob[colL];
#pragma unroll
      for (int e = 0; e < 8; ++e) {
        int r = row0 + e + (hi ? 8 : 0);
        float val = o[e] + bias;
        out[(r * kT + t) * kD + colL] = val;       // fp32 result
        dinp[r * kD + colL] = f2bf(val);           // feed back as next input
      }
    }
    gbar(cnt, gen);
  }
}

// ---------------------------------------------------------------------------
extern "C" void kernel_launch(void* const* d_in, const int* in_sizes, int n_in,
                              void* d_out, int out_size, void* d_ws,
                              size_t ws_size, hipStream_t stream) {
  const float* enc_in = (const float*)d_in[0];
  const float* dec_in = (const float*)d_in[1];
  const float* W0 = (const float*)d_in[2];
  const float* U0 = (const float*)d_in[3];
  const float* b0 = (const float*)d_in[4];
  const float* W1 = (const float*)d_in[5];
  const float* U1 = (const float*)d_in[6];
  const float* b1 = (const float*)d_in[7];
  const float* dW = (const float*)d_in[8];
  const float* dU = (const float*)d_in[9];
  const float* db = (const float*)d_in[10];
  const float* oW = (const float*)d_in[11];
  const float* ob = (const float*)d_in[12];
  float* out = (float*)d_out;

  // workspace carve-up (~5.2 MB total)
  unsigned* bar = (unsigned*)d_ws;
  unsigned short* us = (unsigned short*)((char*)d_ws + 256);
  unsigned short* w0t = us;                 // 1024x128
  unsigned short* u0t = w0t + 131072;       // 1024x256
  unsigned short* w1t = u0t + 262144;
  unsigned short* u1t = w1t + 262144;
  unsigned short* dwt = u1t + 262144;       // 1024x128
  unsigned short* dut = dwt + 131072;       // 1024x256
  unsigned short* owt = dut + 262144;       // 128x256
  unsigned short* h0 = owt + 32768;         // 512x256 bf16
  unsigned short* h1 = h0 + 131072;
  unsigned short* hd = h1 + 131072;
  unsigned short* dinp = hd + 131072;       // 512x128 bf16
  float* fb = (float*)(dinp + 65536);
  float* c0 = fb;                           // 512x256 fp32
  float* c1 = fb + 131072;
  float* cd = fb + 262144;

  lstm_prep<<<2048, 256, 0, stream>>>(W0, U0, W1, U1, dW, dU, oW, dec_in, w0t,
                                      u0t, w1t, u1t, dwt, dut, owt, c0, c1, cd,
                                      h0, h1, hd, dinp, bar);
  lstm_persist<<<NWG, 256, 0, stream>>>(enc_in, b0, b1, db, ob, w0t, u0t, w1t,
                                        u1t, dwt, dut, owt, c0, c1, cd, h0, h1,
                                        hd, dinp, bar, out);
}
